// RadixAttention_46359876993481
// MI455X (gfx1250) — matrix operand order repaired and checked by reading, not compiled, not run on hardware
//
#include <hip/hip_runtime.h>
#include <hip/hip_bf16.h>

typedef __attribute__((ext_vector_type(16))) _Float16 v16h;
typedef __attribute__((ext_vector_type(8)))  _Float16 half8;
typedef __attribute__((ext_vector_type(8)))  float    v8f;
typedef __attribute__((ext_vector_type(4)))  float    f32x4;

// Problem constants (fixed by harness)
constexpr int B_  = 4;
constexpr int E_  = 512;
constexpr int P_  = 1536;
constexpr int H_  = 32;
constexpr int HK_ = 8;
constexpr int D_  = 128;
constexpr int G_  = 4;            // GQA group size H/Hk
constexpr int S_  = P_ + E_;      // 2048
constexpr float SCALE_ = 0.08838834764831845f;

constexpr int CHUNK = 32;         // keys per flash iteration (WMMA K dim for PV)
constexpr int KSTR  = D_ + 8;     // kbuf row stride in halves  (272 B rows, 16B-multiple)
constexpr int VSTR  = CHUNK + 8;  // vtbuf row stride in halves (80 B rows, 16B-multiple)
constexpr int PSTR  = CHUNK + 8;  // pbuf row stride in halves  (80 B rows, 16B-multiple)

__global__ __launch_bounds__(128)
void radix_extend_fa_wmma(const float* __restrict__ q,
                          const float* __restrict__ k,
                          const float* __restrict__ v,
                          const float* __restrict__ kc,
                          const float* __restrict__ vc,
                          float* __restrict__ out) {
  // 16B alignment required: interiors are accessed as half8 (ds_*_b128)
  __shared__ __align__(16) _Float16 kbuf[CHUNK * KSTR];    // K chunk, [s][d], f16
  __shared__ __align__(16) _Float16 vtbuf[D_ * VSTR];      // V chunk transposed, [d][s], f16
  __shared__ __align__(16) _Float16 pbuf[4 * 16 * PSTR];   // per-wave P tile, C->A relayout

  const int tid  = threadIdx.x;
  const int lane = tid & 31;
  const int wv   = tid >> 5;                 // wave id == gi (GQA sub-head)
  const int wg   = blockIdx.x;
  const int et   = wg & 31;                  // e-tile (E/16 = 32)
  const int kh   = (wg >> 5) & 7;            // kv head
  const int b    = wg >> 8;                  // batch
  const int e0   = et * 16;
  const int h    = kh * G_ + wv;             // q head

  const int n   = lane & 15;                 // N col / A row within tile
  const int sel = lane >> 4;                 // lane-half selector

  // ---------------- Q tile -> 4 A-fragments (f16, scale folded in) ----------
  v16h qa[4];
  {
    const float* qrow = q + (((size_t)(b * E_ + e0 + n)) * H_ + h) * D_;
    const f32x4* qp = (const f32x4*)qrow;
#pragma unroll
    for (int f = 0; f < 4; ++f) {
      const int d0 = 32 * f + 8 * sel;       // elements 0..7  -> d0..d0+7
      f32x4 a0 = qp[d0 / 4], a1 = qp[d0 / 4 + 1];
      f32x4 b0 = qp[(d0 + 16) / 4], b1 = qp[(d0 + 16) / 4 + 1];
#pragma unroll
      for (int j = 0; j < 4; ++j) {
        qa[f][j]      = (_Float16)(a0[j] * SCALE_);
        qa[f][4 + j]  = (_Float16)(a1[j] * SCALE_);
        qa[f][8 + j]  = (_Float16)(b0[j] * SCALE_);
        qa[f][12 + j] = (_Float16)(b1[j] * SCALE_);
      }
    }
  }

  // ---------------- flash state ---------------------------------------------
  v8f o[8];
#pragma unroll
  for (int dt = 0; dt < 8; ++dt) o[dt] = v8f{};
  float m[8], l[8];
#pragma unroll
  for (int i = 0; i < 8; ++i) { m[i] = -1e30f; l[i] = 0.0f; }

  const int nvis    = P_ + e0 + 16;                 // keys < nvis may be visible
  const int nchunks = (nvis + CHUNK - 1) / CHUNK;   // max staged key = nchunks*32-1 <= 2047 < S_

  for (int c = 0; c < nchunks; ++c) {
    const int s0 = c * CHUNK;

    __syncthreads();  // previous chunk fully consumed before overwrite

    // ---- stage K chunk: thread -> (row r, 32-wide d slice), f32->f16 -------
    {
      const int r  = tid >> 2;
      const int dp = (tid & 3) * 32;
      const int s  = s0 + r;                 // provably < S_ (see nchunks note)
      const float* src = (s < P_)
          ? kc + (((size_t)(b * P_ + s)) * HK_ + kh) * D_
          : k  + (((size_t)(b * E_ + (s - P_))) * HK_ + kh) * D_;
      const f32x4* sp = (const f32x4*)(src + dp);
      half8* dst = (half8*)&kbuf[r * KSTR + dp];
#pragma unroll
      for (int u = 0; u < 4; ++u) {
        f32x4 x0 = sp[2 * u], x1 = sp[2 * u + 1];
        half8 hh;
#pragma unroll
        for (int j = 0; j < 4; ++j) {
          hh[j]     = (_Float16)x0[j];
          hh[4 + j] = (_Float16)x1[j];
        }
        dst[u] = hh;
      }
    }

    // ---- stage V chunk transposed: thread owns one d column -----------------
    {
      const int d = tid;
      _Float16 col[CHUNK];
#pragma unroll
      for (int r = 0; r < CHUNK; ++r) {
        const int s = s0 + r;               // provably < S_
        const float* src = (s < P_)
            ? vc + (((size_t)(b * P_ + s)) * HK_ + kh) * D_
            : v  + (((size_t)(b * E_ + (s - P_))) * HK_ + kh) * D_;
        col[r] = (_Float16)src[d];
      }
      half8* dst = (half8*)&vtbuf[d * VSTR];
#pragma unroll
      for (int u = 0; u < 4; ++u) {
        half8 hh;
#pragma unroll
        for (int j = 0; j < 8; ++j) hh[j] = col[8 * u + j];
        dst[u] = hh;
      }
    }

    __syncthreads();  // chunk staged

    // ---- scores: two 16x16 tiles, K-dim = D in 4 slices of 32 ---------------
    v8f sc[2];
#pragma unroll
    for (int t = 0; t < 2; ++t) {
      v8f acc = v8f{};
#pragma unroll
      for (int f = 0; f < 4; ++f) {
        // B frag: lane = col n of tile t; elements j<8 -> d = 32f + 8*sel + j, j>=8 -> d = 32f + 16 + 8*sel + (j-8)
        const half8* kp = (const half8*)&kbuf[(t * 16 + n) * KSTR + 32 * f + 8 * sel];
        half8 b0 = kp[0], b1 = kp[2];
        v16h bf;
#pragma unroll
        for (int j = 0; j < 8; ++j) { bf[j] = b0[j]; bf[8 + j] = b1[j]; }
        acc = __builtin_amdgcn_wmma_f32_16x16x32_f16(
            false, qa[f], false, bf, (short)0, acc, false, false);
      }
      sc[t] = acc;
    }

    // ---- online softmax (rows live per lane-half; reduce over 16 lanes) -----
    const int bb = P_ + e0;
#pragma unroll
    for (int i = 0; i < 8; ++i) {
      const int row = i + 8 * sel;
      const int lim = bb + row;                 // key s visible iff s <= lim
      float x0 = sc[0][i];
      float x1 = sc[1][i];
      if (s0 + n      > lim) x0 = -1e30f;
      if (s0 + 16 + n > lim) x1 = -1e30f;
      float tmax = fmaxf(x0, x1);
#pragma unroll
      for (int off = 1; off < 16; off <<= 1)
        tmax = fmaxf(tmax, __shfl_xor(tmax, off, 32));
      const float mn  = fmaxf(m[i], tmax);
      const float rsc = __expf(m[i] - mn);
      const float p0  = __expf(x0 - mn);
      const float p1  = __expf(x1 - mn);
      float ps = p0 + p1;
#pragma unroll
      for (int off = 1; off < 16; off <<= 1)
        ps += __shfl_xor(ps, off, 32);
      l[i] = l[i] * rsc + ps;
      m[i] = mn;
#pragma unroll
      for (int dt = 0; dt < 8; ++dt) o[dt][i] *= rsc;
      pbuf[(wv * 16 + row) * PSTR + n]      = (_Float16)p0;
      pbuf[(wv * 16 + row) * PSTR + 16 + n] = (_Float16)p1;
    }

    // ---- P: C-layout -> A-layout via per-wave LDS (in-order within wave) ----
    v16h pa;
    {
      const half8* pp0 = (const half8*)&pbuf[(wv * 16 + n) * PSTR + 8 * sel];
      const half8* pp1 = (const half8*)&pbuf[(wv * 16 + n) * PSTR + 16 + 8 * sel];
      half8 a0 = pp0[0], a1 = pp1[0];
#pragma unroll
      for (int j = 0; j < 8; ++j) { pa[j] = a0[j]; pa[8 + j] = a1[j]; }
    }

    // ---- PV: 8 d-tiles of 16 cols, K-dim = 32 keys --------------------------
#pragma unroll
    for (int dt = 0; dt < 8; ++dt) {
      // B frag: lane = d col (dt*16+n); elements j<8 -> s = 8*sel + j, j>=8 -> s = 16 + 8*sel + (j-8)
      const half8* vp = (const half8*)&vtbuf[(dt * 16 + n) * VSTR + 8 * sel];
      half8 b0 = vp[0], b1 = vp[2];
      v16h bf;
#pragma unroll
      for (int j = 0; j < 8; ++j) { bf[j] = b0[j]; bf[8 + j] = b1[j]; }
      o[dt] = __builtin_amdgcn_wmma_f32_16x16x32_f16(
          false, pa, false, bf, (short)0, o[dt], false, false);
    }
  }

  // ---------------- epilogue: normalize and store ---------------------------
#pragma unroll
  for (int i = 0; i < 8; ++i) {
    const int row = i + 8 * sel;
    const float inv = 1.0f / l[i];
    float* orow = out + ((size_t)(b * E_ + e0 + row)) * (H_ * D_) + h * D_;
#pragma unroll
    for (int dt = 0; dt < 8; ++dt)
      orow[dt * 16 + n] = o[dt][i] * inv;
  }
}

extern "C" void kernel_launch(void* const* d_in, const int* in_sizes, int n_in,
                              void* d_out, int out_size, void* d_ws, size_t ws_size,
                              hipStream_t stream) {
  (void)in_sizes; (void)n_in; (void)out_size; (void)d_ws; (void)ws_size;
  const float* q  = (const float*)d_in[0];
  const float* k  = (const float*)d_in[1];
  const float* v  = (const float*)d_in[2];
  const float* kc = (const float*)d_in[3];
  const float* vc = (const float*)d_in[4];
  float* out = (float*)d_out;

  dim3 grid(B_ * HK_ * (E_ / 16));   // 1024 workgroups: (b, kv-head, 16-row e-tile)
  dim3 block(128);                   // 4 wave32s: one GQA sub-head each
  radix_extend_fa_wmma<<<grid, block, 0, stream>>>(q, k, v, kc, vc, out);
}
